// Receiver_17961553232209
// MI455X (gfx1250) — compile-verified, matrix operand order checked by reading, not run on hardware
//
#include <hip/hip_runtime.h>
#include <hip/hip_bf16.h>

// ---------- CDNA5 WMMA types ----------
typedef __attribute__((ext_vector_type(16))) __bf16 v16bf;
typedef __attribute__((ext_vector_type(8)))  float  v8f;

struct __align__(16) U4 { unsigned x, y, z, w; };
union Frag { v16bf v; U4 u[2]; };

// ---------- sizes ----------
#define Bsz 256
#define Tsz 64
#define Vsz 4096
#define Esz 256
#define Hsz 512
#define G4H 2048              // 4*H
#define MROWS 16384           // B*T

// ---------- workspace byte offsets ----------
#define XG_OFF   ((size_t)0)                         // x_gates f32 [16384,2048]  = 134217728 B
#define EMBT_OFF ((size_t)134217728)                 // embT bf16  [256,4096]     =   2097152 B
#define WIH_OFF  ((size_t)136314880)                 // W_ih bf16  [2048,256]     =   1048576 B
#define WHH_OFF  ((size_t)137363456)                 // W_hh bf16  [2048,512]     =   2097152 B
#define EMBB_OFF ((size_t)139460608)                 // emb bf16   [16384,256]    =   8388608 B
#define H0_OFF   ((size_t)147849216)                 // h bf16 ping [256,512]     =    262144 B
#define H1_OFF   ((size_t)148111360)                 // h bf16 pong [256,512]     =    262144 B
#define C_OFF    ((size_t)148373504)                 // c f32 [256,512]           =    524288 B

// ---------- helpers ----------
// Native bf16 converts: backend emits v_cvt_pk_bf16_f32 on gfx1250.
__device__ __forceinline__ unsigned short f2bf(float x) {
    union { __bf16 h; unsigned short s; } p;
    p.h = (__bf16)x;
    return p.s;
}
__device__ __forceinline__ unsigned pk2(float lo, float hi) {
    union { __bf16 h[2]; unsigned u; } p;
    p.h[0] = (__bf16)lo;
    p.h[1] = (__bf16)hi;
    return p.u;
}
__device__ __forceinline__ float sigmoid_f(float x) {
    return 1.0f / (1.0f + __expf(-x));
}
__device__ __forceinline__ float tanh_f(float x) {
    return 2.0f / (1.0f + __expf(-2.0f * x)) - 1.0f;
}
// bf16 fragment load: rows stored N(or M)-major with K contiguous (N×K storage)
__device__ __forceinline__ void load_frag_bf(Frag& f, const unsigned short* __restrict__ base,
                                             int ld, int row0, int kb, int lo, int hi) {
    const unsigned short* p = base + (size_t)(row0 + lo) * ld + kb + (hi << 3);
    f.u[0] = *(const U4*)(p);
    f.u[1] = *(const U4*)(p + 16);
}

// ---------- prep kernels ----------
__global__ void zero_state(unsigned short* __restrict__ h0, float* __restrict__ c) {
    int i = blockIdx.x * blockDim.x + threadIdx.x;   // 131072 elements
    if (i < Bsz * Hsz) { h0[i] = 0; c[i] = 0.0f; }
}
__global__ void convert_bf16(const float* __restrict__ src, unsigned short* __restrict__ dst, int n) {
    int i = blockIdx.x * blockDim.x + threadIdx.x;
    if (i < n) dst[i] = f2bf(src[i]);
}
// embedding [V,E] -> embT bf16 [E,V]
__global__ void transpose_emb(const float* __restrict__ emb, unsigned short* __restrict__ out) {
    int i = blockIdx.x * blockDim.x + threadIdx.x;   // over E*V
    if (i < Esz * Vsz) {
        int v = i & (Vsz - 1);
        int e = i >> 12;
        out[i] = f2bf(emb[(size_t)v * Esz + e]);
    }
}

// ---------- GEMM1: emb = messages[16384,4096] @ embedding[4096,256] ----------
// wave tile: 16(M) x 128(N), full K; messages converted f32->bf16 in-register.
// A f32 loads software-pipelined one k-step ahead; sched_barrier(0) pins all
// 16 B b128 loads (distinct regs) ahead of the WMMA chain -> partial waits.
__global__ void __launch_bounds__(256) gemm_emb(const float* __restrict__ A,
                                                const unsigned short* __restrict__ Bt, // [256,4096] bf16
                                                float* __restrict__ Cf,                // d_out emb region
                                                unsigned short* __restrict__ Cb) {     // emb bf16
    const int lane = threadIdx.x & 31;
    const int lo = lane & 15, hi = lane >> 4;
    const int wid = blockIdx.x * 8 + (threadIdx.x >> 5);  // 0..2047
    const int m0 = (wid >> 1) * 16;
    const int n0 = (wid & 1) * 128;

    v8f acc[8];
    const v8f vz = {0.f, 0.f, 0.f, 0.f, 0.f, 0.f, 0.f, 0.f};
#pragma unroll
    for (int i = 0; i < 8; i++) acc[i] = vz;

    const float* arow = A + (size_t)(m0 + lo) * Vsz;
    const int kl = hi << 3;

    // prologue: first A f32 tile
    float4 a0 = *(const float4*)(arow + kl);
    float4 a1 = *(const float4*)(arow + kl + 4);
    float4 a2 = *(const float4*)(arow + kl + 16);
    float4 a3 = *(const float4*)(arow + kl + 20);

    for (int kb = 0; kb < Vsz; kb += 32) {
        __builtin_prefetch(arow + kb + 512, 0, 1);    // global_prefetch_b8, 2KB ahead
        Frag fa;
        fa.u[0] = U4{pk2(a0.x, a0.y), pk2(a0.z, a0.w), pk2(a1.x, a1.y), pk2(a1.z, a1.w)};
        fa.u[1] = U4{pk2(a2.x, a2.y), pk2(a2.z, a2.w), pk2(a3.x, a3.y), pk2(a3.z, a3.w)};

        // pipeline next A tile (redundant reload of last tile on final iter)
        const int kn = (kb + 32 < Vsz) ? (kb + 32 + kl) : (kb + kl);
        a0 = *(const float4*)(arow + kn);
        a1 = *(const float4*)(arow + kn + 4);
        a2 = *(const float4*)(arow + kn + 16);
        a3 = *(const float4*)(arow + kn + 20);

        // all 8 B fragments issued before any WMMA (kept in distinct regs)
        Frag fb[8];
#pragma unroll
        for (int nt = 0; nt < 8; nt++)
            load_frag_bf(fb[nt], Bt, Vsz, n0 + nt * 16, kb, lo, hi);

        __builtin_amdgcn_sched_barrier(0);

#pragma unroll
        for (int nt = 0; nt < 8; nt++)
            acc[nt] = __builtin_amdgcn_wmma_f32_16x16x32_bf16(
                false, fa.v, false, fb[nt].v, (short)0, acc[nt], false, false);
    }
#pragma unroll
    for (int nt = 0; nt < 8; nt++) {
        const int n = n0 + nt * 16 + lo;
#pragma unroll
        for (int r = 0; r < 8; r++) {
            const int m = m0 + r + (hi << 3);
            const float v = acc[nt][r];
            Cf[(size_t)m * Esz + n] = v;
            Cb[(size_t)m * Esz + n] = f2bf(v);
        }
    }
}

// ---------- GEMM2: x_gates = emb_bf16[16384,256] @ W_ih^T (+ b_ih + b_hh) ----------
__global__ void __launch_bounds__(256) gemm_xg(const unsigned short* __restrict__ Ab,  // [16384,256]
                                               const unsigned short* __restrict__ Bb,  // [2048,256]
                                               const float* __restrict__ bih,
                                               const float* __restrict__ bhh,
                                               float* __restrict__ XG) {               // [16384,2048]
    const int lane = threadIdx.x & 31;
    const int lo = lane & 15, hi = lane >> 4;
    const int wid = blockIdx.x * 8 + (threadIdx.x >> 5);  // 0..16383
    const int m0 = (wid >> 4) * 16;
    const int n0 = (wid & 15) * 128;

    v8f acc[8];
    const v8f vz = {0.f, 0.f, 0.f, 0.f, 0.f, 0.f, 0.f, 0.f};
#pragma unroll
    for (int i = 0; i < 8; i++) acc[i] = vz;

    Frag fa;
    load_frag_bf(fa, Ab, Esz, m0, 0, lo, hi);

#pragma unroll
    for (int kb = 0; kb < Esz; kb += 32) {
        Frag fb[8];
#pragma unroll
        for (int nt = 0; nt < 8; nt++)
            load_frag_bf(fb[nt], Bb, Esz, n0 + nt * 16, kb, lo, hi);

        const Frag fac = fa;
        if (kb + 32 < Esz) load_frag_bf(fa, Ab, Esz, m0, kb + 32, lo, hi);

        __builtin_amdgcn_sched_barrier(0);

#pragma unroll
        for (int nt = 0; nt < 8; nt++)
            acc[nt] = __builtin_amdgcn_wmma_f32_16x16x32_bf16(
                false, fac.v, false, fb[nt].v, (short)0, acc[nt], false, false);
    }
#pragma unroll
    for (int nt = 0; nt < 8; nt++) {
        const int n = n0 + nt * 16 + lo;
        const float bias = bih[n] + bhh[n];
#pragma unroll
        for (int r = 0; r < 8; r++) {
            const int m = m0 + r + (hi << 3);
            XG[(size_t)m * G4H + n] = acc[nt][r] + bias;
        }
    }
}

// ---------- LSTM step (fused GEMM + gates), one launch per t ----------
// wave tile: 16 batch rows x 32 hidden cols, computing all 4 gate blocks
__global__ void __launch_bounds__(256) lstm_step(const unsigned short* __restrict__ Hin, // [256,512] bf16
                                                 const unsigned short* __restrict__ Whh, // [2048,512] bf16
                                                 const float* __restrict__ XG,           // [16384,2048]
                                                 float* __restrict__ Cst,                // [256,512]
                                                 unsigned short* __restrict__ Hout,      // bf16
                                                 float* __restrict__ Hf,                 // f32 -> d_out h
                                                 int t) {
    const int lane = threadIdx.x & 31;
    const int lo = lane & 15, hi = lane >> 4;
    const int wid = blockIdx.x * 8 + (threadIdx.x >> 5);  // 0..255
    const int m0 = (wid >> 4) * 16;                       // batch tile
    const int j0 = (wid & 15) * 32;                       // hidden-col slice

    v8f acc[8];                                           // [gate][ns] -> gate*2+ns
    const v8f vz = {0.f, 0.f, 0.f, 0.f, 0.f, 0.f, 0.f, 0.f};
#pragma unroll
    for (int i = 0; i < 8; i++) acc[i] = vz;

    Frag fa;
    load_frag_bf(fa, Hin, Hsz, m0, 0, lo, hi);

    for (int kb = 0; kb < Hsz; kb += 32) {
        Frag fb[8];
#pragma unroll
        for (int gate = 0; gate < 4; gate++)
#pragma unroll
            for (int ns = 0; ns < 2; ns++)
                load_frag_bf(fb[gate * 2 + ns], Whh, Hsz,
                             gate * Hsz + j0 + ns * 16, kb, lo, hi);

        const Frag fac = fa;
        if (kb + 32 < Hsz) load_frag_bf(fa, Hin, Hsz, m0, kb + 32, lo, hi);

        __builtin_amdgcn_sched_barrier(0);

#pragma unroll
        for (int q = 0; q < 8; q++)
            acc[q] = __builtin_amdgcn_wmma_f32_16x16x32_bf16(
                false, fac.v, false, fb[q].v, (short)0, acc[q], false, false);
    }
#pragma unroll
    for (int ns = 0; ns < 2; ns++) {
        const int j = j0 + ns * 16 + lo;
#pragma unroll
        for (int r = 0; r < 8; r++) {
            const int b = m0 + r + (hi << 3);
            const size_t xb = ((size_t)b * Tsz + t) * G4H;
            const float iv = acc[0 * 2 + ns][r] + XG[xb + 0 * Hsz + j];
            const float fv = acc[1 * 2 + ns][r] + XG[xb + 1 * Hsz + j];
            const float gv = acc[2 * 2 + ns][r] + XG[xb + 2 * Hsz + j];
            const float ov = acc[3 * 2 + ns][r] + XG[xb + 3 * Hsz + j];
            const float ig = sigmoid_f(iv);
            const float fg = sigmoid_f(fv);
            const float gg = tanh_f(gv);
            const float og = sigmoid_f(ov);
            const size_t ci = (size_t)b * Hsz + j;
            const float c = fg * Cst[ci] + ig * gg;
            Cst[ci] = c;
            const float h = og * tanh_f(c);
            Hf[ci] = h;
            Hout[ci] = f2bf(h);
        }
    }
}

// ---------- launch ----------
extern "C" void kernel_launch(void* const* d_in, const int* in_sizes, int n_in,
                              void* d_out, int out_size, void* d_ws, size_t ws_size,
                              hipStream_t stream) {
    const float* messages  = (const float*)d_in[0];
    const float* embedding = (const float*)d_in[1];
    const float* W_ih      = (const float*)d_in[2];
    const float* W_hh      = (const float*)d_in[3];
    const float* b_ih      = (const float*)d_in[4];
    const float* b_hh      = (const float*)d_in[5];

    float* out_h   = (float*)d_out;                 // [256,512]
    float* out_emb = out_h + (size_t)Bsz * Hsz;     // [256,64,256]

    char* ws = (char*)d_ws;
    float*          xg   = (float*)(ws + XG_OFF);
    unsigned short* embT = (unsigned short*)(ws + EMBT_OFF);
    unsigned short* wihb = (unsigned short*)(ws + WIH_OFF);
    unsigned short* whhb = (unsigned short*)(ws + WHH_OFF);
    unsigned short* embb = (unsigned short*)(ws + EMBB_OFF);
    unsigned short* h0   = (unsigned short*)(ws + H0_OFF);
    unsigned short* h1   = (unsigned short*)(ws + H1_OFF);
    float*          cst  = (float*)(ws + C_OFF);

    zero_state<<<(Bsz * Hsz + 255) / 256, 256, 0, stream>>>(h0, cst);
    transpose_emb<<<(Esz * Vsz + 255) / 256, 256, 0, stream>>>(embedding, embT);
    convert_bf16<<<(G4H * Esz + 255) / 256, 256, 0, stream>>>(W_ih, wihb, G4H * Esz);
    convert_bf16<<<(G4H * Hsz + 255) / 256, 256, 0, stream>>>(W_hh, whhb, G4H * Hsz);

    // 2048 waves (1024 m-tiles x 2 n-strips), 8 waves/block
    gemm_emb<<<256, 256, 0, stream>>>(messages, embT, out_emb, embb);
    // 16384 waves (1024 m-tiles x 16 n-strips)
    gemm_xg<<<2048, 256, 0, stream>>>(embb, wihb, b_ih, b_hh, xg);

    for (int t = 0; t < Tsz; t++) {
        unsigned short* hin  = (t & 1) ? h1 : h0;
        unsigned short* hout = (t & 1) ? h0 : h1;
        lstm_step<<<32, 256, 0, stream>>>(hin, whhb, xg, cst, hout, out_h, t);
    }
}
